// EmformerAttention_18485539242029
// MI455X (gfx1250) — matrix operand order repaired and checked by reading, not compiled, not run on hardware
//
#include <hip/hip_runtime.h>
#include <hip/hip_bf16.h>
#include <math.h>

typedef __bf16 bf16;
typedef __attribute__((ext_vector_type(16))) __bf16 v16bf;
typedef __attribute__((ext_vector_type(8)))  float  v8f;

#define NHEADS  8
#define DH      64
#define DMODEL  512
#define BATCH   4
#define UU      2048
#define RR      64
#define SSUM    64
#define MMEM    64
#define QLEN    2176          // R + U + S
#define KVLEN   2176          // M + R + U
#define ROWS    (QLEN*BATCH)  // 8704
#define NEGINF  (-100000000.0f)
#define TLD     72            // LDS tile leading dim (bf16 elems): 144B rows, 16B aligned

// ---- WMMA fragment loaders ------------------------------------------------
// A (16x32 bf16): lane m=lane&15, h=lane>>4; elems 0..7 -> k=koff+8h.., 8..15 -> k=koff+8h+16..
__device__ __forceinline__ v16bf load_frag_a(const bf16* rowbase, int koff, int h) {
  union { v16bf v; int4 q[2]; } u;
  u.q[0] = *reinterpret_cast<const int4*>(rowbase + koff + 8*h);
  u.q[1] = *reinterpret_cast<const int4*>(rowbase + koff + 8*h + 16);
  return u.v;
}
// B (32x16 bf16): lane n=lane&15, h=lane>>4; elems 0..7 -> k=8h.., 8..15 -> k=8h+16.. (p = column n at k=0, row stride ld)
__device__ __forceinline__ v16bf load_frag_b(const bf16* p, int ld, int h) {
  v16bf v;
#pragma unroll
  for (int s = 0; s < 8; ++s) {
    v[s] = p[(8*h + s) * ld];
    v[8 + s] = p[(16 + 8*h + s) * ld];
  }
  return v;
}

// One wave computes a 16x64 fp32 tile: A(16x64) x B(64x64), K=64 in two 32-chunks.
__device__ __forceinline__ void gemm_wave_16x64(const bf16* a_base, int a_ld,
                                                const bf16* b_base, int b_ld,
                                                v8f acc[4]) {
  const int lane = threadIdx.x & 31;
  const int m = lane & 15, h = lane >> 4;
  const bf16* arow = a_base + m * a_ld;
#pragma unroll
  for (int c = 0; c < 2; ++c) {
    v16bf a = load_frag_a(arow, 32*c, h);
    const bf16* bk = b_base + (32*c) * b_ld;
#pragma unroll
    for (int nt = 0; nt < 4; ++nt) {
      v16bf bb = load_frag_b(bk + nt*16 + m, b_ld, h);
      acc[nt] = __builtin_amdgcn_wmma_f32_16x16x32_bf16(
          false, a, false, bb, (short)0, acc[nt], false, false);
    }
  }
}

// pack 4 floats -> 4 bf16 (8 bytes)
__device__ __forceinline__ uint2 pack4bf(float4 f) {
  union { bf16 h[4]; uint2 u; } cv;
  cv.h[0] = (bf16)f.x; cv.h[1] = (bf16)f.y; cv.h[2] = (bf16)f.z; cv.h[3] = (bf16)f.w;
  return cv.u;
}

// ---- Kernel 1/2: fused gather + projection GEMM ---------------------------
// mode 0: q_in = [rctx|utt|summary] @ w_q (+b_q) * scale -> out0 = qs[row][n]
// mode 1: kv_in = [mem|rctx|utt] @ w_kv (+b_kv):
//         n<512 -> K, stored TRANSPOSED per head: ksT[((b*8+h)*64+d)*KVLEN + kv]
//         n>=512 -> V: vs[row][n-512]
__global__ __launch_bounds__(128) void proj_kernel(
    const float* __restrict__ utter, const float* __restrict__ rctx,
    const float* __restrict__ summ,  const float* __restrict__ memo,
    const float* __restrict__ w, const float* __restrict__ bias,
    bf16* __restrict__ out0, bf16* __restrict__ out1,
    int n_cols, int mode, float scale) {
  __shared__ __attribute__((aligned(16))) bf16 lA[64*TLD];
  __shared__ __attribute__((aligned(16))) bf16 lB[64*TLD];
  const int tid = threadIdx.x, wv = tid >> 5, lane = tid & 31;
  const int m = lane & 15, h = lane >> 4;
  const int col_tiles = n_cols >> 6;
  const int rt = blockIdx.x / col_tiles, ct = blockIdx.x % col_tiles;

  v8f acc[4] = {};
  for (int kb = 0; kb < 8; ++kb) {
    // stage A (gathered input, fp32 -> bf16), float4-vectorized
    for (int id = tid; id < 1024; id += 128) {
      int r = id >> 4, c4 = id & 15;
      int row = rt*64 + r;
      int pos = row >> 2, b = row & 3;
      const float* src;
      if (mode == 0) {
        if (pos < RR)            src = rctx + (size_t)(pos*BATCH + b)*DMODEL;
        else if (pos < RR + UU)  src = utter + (size_t)((pos-RR)*BATCH + b)*DMODEL;
        else                     src = summ + (size_t)((pos-RR-UU)*BATCH + b)*DMODEL;
      } else {
        if (pos < MMEM)            src = memo + (size_t)(pos*BATCH + b)*DMODEL;
        else if (pos < MMEM + RR)  src = rctx + (size_t)((pos-MMEM)*BATCH + b)*DMODEL;
        else                       src = utter + (size_t)((pos-MMEM-RR)*BATCH + b)*DMODEL;
      }
      float4 f = *reinterpret_cast<const float4*>(src + kb*64 + c4*4);
      *reinterpret_cast<uint2*>(&lA[r*TLD + c4*4]) = pack4bf(f);
    }
    // stage B (weights, fp32 -> bf16)
    for (int id = tid; id < 1024; id += 128) {
      int r = id >> 4, c4 = id & 15;
      float4 f = *reinterpret_cast<const float4*>(
          w + (size_t)(kb*64 + r)*n_cols + ct*64 + c4*4);
      *reinterpret_cast<uint2*>(&lB[r*TLD + c4*4]) = pack4bf(f);
    }
    __syncthreads();
    gemm_wave_16x64(lA + (wv*16)*TLD, TLD, lB, TLD, acc);
    __syncthreads();
  }
#pragma unroll
  for (int nt = 0; nt < 4; ++nt) {
    int n = ct*64 + nt*16 + m;
    float bv = bias[n];
#pragma unroll
    for (int i = 0; i < 8; ++i) {
      int row = rt*64 + wv*16 + i + 8*h;
      float v = (acc[nt][i] + bv) * scale;
      if (mode == 0) {
        out0[(size_t)row*DMODEL + n] = (bf16)v;
      } else {
        int kv = row >> 2, b = row & 3;
        if (n < DMODEL) {
          int hd = n >> 6, d = n & 63;                     // head, dim
          out0[((size_t)(b*NHEADS + hd)*DH + d)*KVLEN + kv] = (bf16)v;  // K^T
        } else {
          out1[(size_t)row*DMODEL + (n - DMODEL)] = (bf16)v;            // V
        }
      }
    }
  }
}

// ---- Kernel 3: flash-style attention per (b, head, 64-query tile) ---------
__global__ __launch_bounds__(128) void attn_kernel(
    const bf16* __restrict__ qs, const bf16* __restrict__ ksT,
    const bf16* __restrict__ vs, const unsigned char* __restrict__ amask,
    const int* __restrict__ lengths, bf16* __restrict__ ctx) {
  __shared__ __attribute__((aligned(16))) bf16  kT[64*TLD];  // [d][kj]
  __shared__ __attribute__((aligned(16))) bf16  vt[64*TLD];  // [kj][d]
  __shared__ __attribute__((aligned(16))) bf16  pt[64*TLD];  // P tile bf16
  __shared__ __attribute__((aligned(16))) float st[64*TLD];  // S tile fp32
  __shared__ float rscale[64];

  const int tid = threadIdx.x, wv = tid >> 5, lane = tid & 31;
  const int m = lane & 15, h = lane >> 4;
  const int qt = blockIdx.x % 34;
  const int hh = (blockIdx.x / 34) % NHEADS;
  const int b  = blockIdx.x / (34 * NHEADS);
  const int qi0 = qt * 64;
  const int limit = (MMEM + RR) + lengths[b];  // pad mask: kj >= 128 + len[b]

  // Q fragments (pre-scaled bf16) held in registers for the whole KV sweep
  const bf16* qrow = qs + ((size_t)(qi0 + wv*16 + m)*BATCH + b)*DMODEL + hh*DH;
  const v16bf qa0 = load_frag_a(qrow, 0, h);
  const v16bf qa1 = load_frag_a(qrow, 32, h);

  const bf16* ktb = ksT + (size_t)(b*NHEADS + hh)*DH*KVLEN;  // [d][kv]
  const size_t vrow0 = (size_t)b*DMODEL + hh*DH;             // + kv*BATCH*DMODEL

  // per-row softmax state: lane pair (2r, 2r+1) shares row r = tid>>1
  const int srow_i = tid >> 1, shalf = tid & 1;
  const int qg = qi0 + srow_i;

  v8f acc[4] = {};
  float mi = -INFINITY, li = 0.0f;

  for (int kb = 0; kb < 34; ++kb) {
    const int kj0 = kb * 64;
    // stage K^T and V tiles with 16B copies (8 int4 per thread)
    for (int id = tid; id < 512; id += 128) {
      int rr = id >> 3, c8 = id & 7;
      *reinterpret_cast<int4*>(&kT[rr*TLD + c8*8]) =
          *reinterpret_cast<const int4*>(ktb + (size_t)rr*KVLEN + kj0 + c8*8);
      *reinterpret_cast<int4*>(&vt[rr*TLD + c8*8]) =
          *reinterpret_cast<const int4*>(vs + vrow0 + (size_t)(kj0 + rr)*BATCH*DMODEL + c8*8);
    }
    __syncthreads();

    // S = Q x K^T  (16x64 per wave)
    v8f sacc[4] = {};
#pragma unroll
    for (int c = 0; c < 2; ++c) {
      v16bf a = (c == 0) ? qa0 : qa1;
      const bf16* bk = kT + (32*c) * TLD;
#pragma unroll
      for (int nt = 0; nt < 4; ++nt) {
        v16bf bb = load_frag_b(bk + nt*16 + m, TLD, h);
        sacc[nt] = __builtin_amdgcn_wmma_f32_16x16x32_bf16(
            false, a, false, bb, (short)0, sacc[nt], false, false);
      }
    }
#pragma unroll
    for (int nt = 0; nt < 4; ++nt)
#pragma unroll
      for (int i = 0; i < 8; ++i)
        st[(wv*16 + i + 8*h)*TLD + nt*16 + m] = sacc[nt][i];
    __syncthreads();

    // masking + online softmax: lane pair per row, 32 cols each, shfl_xor combine
    {
      float* srow = st + srow_i*TLD + shalf*32;
      bf16*  prow = pt + srow_i*TLD + shalf*32;
      const unsigned int* mrow = reinterpret_cast<const unsigned int*>(
          amask + (size_t)qg * KVLEN + kj0 + shalf*32);
      float mloc = -INFINITY;
#pragma unroll
      for (int w8 = 0; w8 < 8; ++w8) {
        unsigned int mw = mrow[w8];
#pragma unroll
        for (int j = 0; j < 4; ++j) {
          int c = w8*4 + j;
          float sv = srow[c];
          bool mskd = ((mw >> (8*j)) & 0xffu) || ((kj0 + shalf*32 + c) >= limit);
          sv = mskd ? NEGINF : sv;   // NEG_INF finite -> uniform rows like ref
          srow[c] = sv;
          mloc = fmaxf(mloc, sv);
        }
      }
      float mnew = fmaxf(mi, fmaxf(mloc, __shfl_xor(mloc, 1)));
      float sc = __expf(mi - mnew);  // first tile: exp(-inf)=0
      float ls = 0.0f;
#pragma unroll
      for (int c = 0; c < 32; ++c) {
        float p = __expf(srow[c] - mnew);
        prow[c] = (bf16)p;
        ls += p;
      }
      ls += __shfl_xor(ls, 1);
      li = li * sc + ls;
      mi = mnew;
      if (shalf == 0) rscale[srow_i] = sc;
    }
    __syncthreads();

    // rescale accumulator by exp(m_old - m_new), then acc += P x V
    {
      float f[8];
#pragma unroll
      for (int i = 0; i < 8; ++i) f[i] = rscale[wv*16 + i + 8*h];
#pragma unroll
      for (int nt = 0; nt < 4; ++nt)
#pragma unroll
        for (int i = 0; i < 8; ++i) acc[nt][i] *= f[i];
    }
#pragma unroll
    for (int c = 0; c < 2; ++c) {
      v16bf a = load_frag_a(pt + (wv*16 + m)*TLD, 32*c, h);
      const bf16* bk = vt + (32*c) * TLD;
#pragma unroll
      for (int nt = 0; nt < 4; ++nt) {
        v16bf bb = load_frag_b(bk + nt*16 + m, TLD, h);
        acc[nt] = __builtin_amdgcn_wmma_f32_16x16x32_bf16(
            false, a, false, bb, (short)0, acc[nt], false, false);
      }
    }
    __syncthreads();
  }

  if (shalf == 0) rscale[srow_i] = 1.0f / li;   // li > 0 always (NEG_INF finite)
  __syncthreads();
#pragma unroll
  for (int nt = 0; nt < 4; ++nt)
#pragma unroll
    for (int i = 0; i < 8; ++i) {
      int r = wv*16 + i + 8*h;
      float v = acc[nt][i] * rscale[r];
      ctx[((size_t)(qi0 + r)*BATCH + b)*DMODEL + hh*DH + nt*16 + m] = (bf16)v;
    }
}

// ---- Kernel 4: output projection + split/clip -----------------------------
__global__ __launch_bounds__(128) void outproj_kernel(
    const bf16* __restrict__ ctxb, const float* __restrict__ w_out,
    const float* __restrict__ b_out, float* __restrict__ out) {
  __shared__ __attribute__((aligned(16))) bf16 lA[64*TLD];
  __shared__ __attribute__((aligned(16))) bf16 lB[64*TLD];
  const int tid = threadIdx.x, wv = tid >> 5, lane = tid & 31;
  const int m = lane & 15, h = lane >> 4;
  const int rt = blockIdx.x >> 3, ct = blockIdx.x & 7;

  v8f acc[4] = {};
  for (int kb = 0; kb < 8; ++kb) {
    for (int id = tid; id < 512; id += 128) {   // A: bf16 straight 16B copies
      int r = id >> 3, c8 = id & 7;
      *reinterpret_cast<int4*>(&lA[r*TLD + c8*8]) =
          *reinterpret_cast<const int4*>(ctxb + (size_t)(rt*64 + r)*DMODEL + kb*64 + c8*8);
    }
    for (int id = tid; id < 1024; id += 128) {  // B: fp32 -> bf16
      int r = id >> 4, c4 = id & 15;
      float4 f = *reinterpret_cast<const float4*>(
          w_out + (size_t)(kb*64 + r)*DMODEL + ct*64 + c4*4);
      *reinterpret_cast<uint2*>(&lB[r*TLD + c4*4]) = pack4bf(f);
    }
    __syncthreads();
    gemm_wave_16x64(lA + (wv*16)*TLD, TLD, lB, TLD, acc);
    __syncthreads();
  }
#pragma unroll
  for (int nt = 0; nt < 4; ++nt) {
    int n = ct*64 + nt*16 + m;
    float bv = b_out[n];
#pragma unroll
    for (int i = 0; i < 8; ++i) {
      int row = rt*64 + wv*16 + i + 8*h;
      int q = row >> 2, b = row & 3;
      float v = acc[nt][i] + bv;
      if (q < QLEN - SSUM) {
        out[(size_t)row*DMODEL + n] = v;                  // out_rcu, flat (q,b,d)
      } else if (q < QLEN - 1) {                          // out_mem[:-1], clipped
        v = fminf(fmaxf(v, -10.0f), 10.0f);
        out[(size_t)(QLEN - SSUM)*BATCH*DMODEL +
            (size_t)((q - (QLEN - SSUM))*BATCH + b)*DMODEL + n] = v;
      } // last summary row dropped
    }
  }
}

extern "C" void kernel_launch(void* const* d_in, const int* in_sizes, int n_in,
                              void* d_out, int out_size, void* d_ws, size_t ws_size,
                              hipStream_t stream) {
  const float* utter = (const float*)d_in[0];
  const int*   lens  = (const int*)d_in[1];
  const float* rctx  = (const float*)d_in[2];
  const float* summ  = (const float*)d_in[3];
  const float* memo  = (const float*)d_in[4];
  const unsigned char* amask = (const unsigned char*)d_in[5];
  const float* w_q   = (const float*)d_in[6];
  const float* b_q   = (const float*)d_in[7];
  const float* w_kv  = (const float*)d_in[8];
  const float* b_kv  = (const float*)d_in[9];
  const float* w_out = (const float*)d_in[10];
  const float* b_out = (const float*)d_in[11];

  const size_t plane = (size_t)ROWS * DMODEL;  // 8704*512 bf16 elems
  bf16* qs  = (bf16*)d_ws;
  bf16* ksT = qs + plane;   // K transposed per (b,h): [b][h][d][kv]
  bf16* vs  = ksT + plane;
  bf16* cx  = vs + plane;   // total ~34 MB of workspace

  dim3 blk(128);
  // Q projection: 136 row tiles x 8 col tiles, scale = dh^-0.5 = 0.125
  proj_kernel<<<dim3(136*8), blk, 0, stream>>>(utter, rctx, summ, memo,
                                               w_q, b_q, qs, nullptr, 512, 0, 0.125f);
  // KV projection: N=1024, K -> transposed ksT, V -> vs
  proj_kernel<<<dim3(136*16), blk, 0, stream>>>(utter, rctx, summ, memo,
                                                w_kv, b_kv, ksT, vs, 1024, 1, 1.0f);
  // Attention: B * NHEADS * (2176/64) blocks
  attn_kernel<<<dim3(BATCH*NHEADS*34), blk, 0, stream>>>(qs, ksT, vs, amask, lens, cx);
  // Output projection + split/clip
  outproj_kernel<<<dim3(136*8), blk, 0, stream>>>(cx, w_out, b_out, (float*)d_out);
}
